// SparseAttention_12515534701248
// MI455X (gfx1250) — compile-verified, matrix operand order checked by reading, not run on hardware
//
#include <hip/hip_runtime.h>
#include <hip/hip_bf16.h>

typedef __attribute__((ext_vector_type(16))) _Float16 v16h;
typedef __attribute__((ext_vector_type(8)))  float    v8f;

#define B_      2
#define H_      16
#define S_      2048
#define D_      128
#define HALF_W  102            // w = int(2048*0.1)=204, half = 102
#define NEGV    (-1.0e9f)
#define SCOLS   256            // >= max window tiles (15) * 16
#define VT_STR  34             // padded (even) LDS stride for transposed V

// Build a 16-bit WMMA fragment per CDNA5 ISA 7.12.2 (16-bit A 16x32 layout).
// lane m/n = lane&15, hi = lane>>4. Element pairs are consecutive K values,
// loaded as dword pairs from LDS. Used for A (row-major over K) and for B
// operands staged so that the fragment's K runs contiguously in LDS.
__device__ __forceinline__ v16h frag16(const _Float16* base, int stride,
                                       int row, int hi, int k0) {
  union { v16h h; unsigned int u[8]; } f;
  const unsigned int* p =
      (const unsigned int*)(base + (size_t)row * stride + k0);
#pragma unroll
  for (int v = 0; v < 8; ++v) {
    int k = (v < 4) ? (hi * 8 + v * 2) : (16 + hi * 8 + (v - 4) * 2);
    f.u[v] = p[k >> 1];
  }
  return f.h;
}

__device__ __forceinline__ v8f wmma_f16(v16h a, v16h b, v8f c) {
  return __builtin_amdgcn_wmma_f32_16x16x32_f16(
      /*neg_a=*/false, a, /*neg_b=*/false, b,
      /*c_mod=*/(short)0, c, /*reuse_a=*/false, /*reuse_b=*/false);
}

__global__ __launch_bounds__(128) void sparse_attn_kernel(
    const float* __restrict__ Q, const float* __restrict__ K,
    const float* __restrict__ V, float* __restrict__ ctx,
    float* __restrict__ probs) {
  __shared__ _Float16 Qh[16][D_];          // 4 KB
  __shared__ _Float16 Kst[4][16][D_];      // 16 KB, one K-tile per wave
  __shared__ float    Sc[16][SCOLS];       // 16 KB, scores then probs (f32)
  __shared__ _Float16 Ph[16][SCOLS];       // 8 KB, normalized probs (f16)
  __shared__ _Float16 Vt[D_][VT_STR];      // ~8.5 KB, V chunk transposed

  const int tid  = threadIdx.x;
  const int wave = tid >> 5;
  const int lane = tid & 31;
  const int m    = lane & 15;
  const int hi   = lane >> 4;

  const int qt = blockIdx.x & (S_ / 16 - 1);
  const int bh = blockIdx.x / (S_ / 16);
  const int q0 = qt * 16;

  const float* Qb = Q + (size_t)bh * S_ * D_;
  const float* Kb = K + (size_t)bh * S_ * D_;
  const float* Vb = V + (size_t)bh * S_ * D_;

  // ---- window extent for this 16-row query tile ----
  int klo = q0 - HALF_W;            if (klo < 0) klo = 0;
  int khi = q0 + 15 + HALF_W;       if (khi > S_ - 1) khi = S_ - 1;
  const int t0 = klo >> 4, t1 = khi >> 4;
  const int ntiles = t1 - t0 + 1;   // <= 15
  const int ncols  = ntiles * 16;

  // ---- stage Q tile as f16; zero Ph padding ----
  for (int i = tid; i < 16 * D_ / 4; i += 128) {
    int r = i >> 5, c = (i & 31) * 4;
    const float4 q = *(const float4*)(Qb + (size_t)(q0 + r) * D_ + c);
    Qh[r][c + 0] = (_Float16)q.x; Qh[r][c + 1] = (_Float16)q.y;
    Qh[r][c + 2] = (_Float16)q.z; Qh[r][c + 3] = (_Float16)q.w;
  }
  for (int i = tid; i < 16 * SCOLS; i += 128)
    Ph[i >> 8][i & (SCOLS - 1)] = (_Float16)0.0f;
  __syncthreads();

  // ---- phase 1: scores = QK^T/sqrt(D) with band mask ----
  const float scale = 0.08838834764831845f;   // 1/sqrt(128)
  for (int it = 0; it < 4; ++it) {
    const int tt = t0 + wave + it * 4;        // wave-uniform predicate
    if (tt <= t1) {
      // stage 16x128 K tile into this wave's LDS slab (f16)
      const int r = lane >> 1, cb = (lane & 1) * 64;
      const float4* Kr4 = (const float4*)(Kb + (size_t)(tt * 16 + r) * D_ + cb);
#pragma unroll
      for (int j = 0; j < 16; ++j) {
        const float4 kv = Kr4[j];
        _Float16* d = &Kst[wave][r][cb + j * 4];
        d[0] = (_Float16)kv.x; d[1] = (_Float16)kv.y;
        d[2] = (_Float16)kv.z; d[3] = (_Float16)kv.w;
      }
      // same-wave DS ordering fence before cross-lane LDS reads
      asm volatile("s_wait_dscnt 0" ::: "memory");
      __builtin_amdgcn_wave_barrier();

      v8f acc = {};
#pragma unroll
      for (int d0 = 0; d0 < D_; d0 += 32) {
        v16h a = frag16(&Qh[0][0], D_, m, hi, d0);          // A: Q rows
        v16h b = frag16(&Kst[wave][0][0], D_, m, hi, d0);   // B: K^T (n=m)
        acc = wmma_f16(a, b, acc);
      }
      const int colg = tt * 16 + m;
      const int cloc = (tt - t0) * 16 + m;
#pragma unroll
      for (int r8 = 0; r8 < 8; ++r8) {
        const int rowg = q0 + r8 + hi * 8;
        float s = acc[r8] * scale;
        const int diff = colg - rowg;
        if (diff < -HALF_W || diff > HALF_W) s = NEGV;
        Sc[r8 + hi * 8][cloc] = s;
      }
    }
  }
  __syncthreads();

  // ---- phase 2: row softmax (8 lanes per row, wave32 shuffles) ----
  {
    const int row = tid >> 3, sub = tid & 7;
    float mx = -3.0e38f;
    for (int c = sub; c < ncols; c += 8) mx = fmaxf(mx, Sc[row][c]);
    mx = fmaxf(mx, __shfl_xor(mx, 1, 32));
    mx = fmaxf(mx, __shfl_xor(mx, 2, 32));
    mx = fmaxf(mx, __shfl_xor(mx, 4, 32));
    float sum = 0.0f;
    for (int c = sub; c < ncols; c += 8) {
      const float e = expf(Sc[row][c] - mx);   // masked -> exactly 0
      Sc[row][c] = e; sum += e;
    }
    sum += __shfl_xor(sum, 1, 32);
    sum += __shfl_xor(sum, 2, 32);
    sum += __shfl_xor(sum, 4, 32);
    const float inv = 1.0f / sum;
    for (int c = sub; c < ncols; c += 8) {
      const float p = Sc[row][c] * inv;
      Sc[row][c] = p; Ph[row][c] = (_Float16)p;
    }
  }
  __syncthreads();

  // ---- phase 3: context = P * V, K-chunks of 32, 2 n-tiles per wave ----
  v8f c0 = {}, c1 = {};
  const int n0  = wave * 32;
  const int nkk = (ntiles + 1) >> 1;
  for (int kk = 0; kk < nkk; ++kk) {
    // stage 32x128 V chunk transposed: Vt[col][k]
    for (int i = tid; i < 32 * D_ / 4; i += 128) {
      const int r = i >> 5, c = (i & 31) * 4;
      const int rowg = t0 * 16 + kk * 32 + r;
      float4 vv = make_float4(0.f, 0.f, 0.f, 0.f);
      if (rowg < S_) vv = *(const float4*)(Vb + (size_t)rowg * D_ + c);
      Vt[c + 0][r] = (_Float16)vv.x; Vt[c + 1][r] = (_Float16)vv.y;
      Vt[c + 2][r] = (_Float16)vv.z; Vt[c + 3][r] = (_Float16)vv.w;
    }
    __syncthreads();
    v16h a  = frag16(&Ph[0][0], SCOLS, m, hi, kk * 32);      // A: P rows
    v16h b0 = frag16(&Vt[0][0], VT_STR, n0 + m,      hi, 0); // B: V cols
    v16h b1 = frag16(&Vt[0][0], VT_STR, n0 + 16 + m, hi, 0);
    c0 = wmma_f16(a, b0, c0);
    c1 = wmma_f16(a, b1, c1);
    __syncthreads();
  }

  // ---- write context, non-temporal (written once, never re-read) ----
  // C/D layout: VGPR r -> row r + hi*8, col = n
#pragma unroll
  for (int r8 = 0; r8 < 8; ++r8) {
    const int rowg = q0 + r8 + hi * 8;
    float* orow = ctx + ((size_t)bh * S_ + rowg) * D_;
    __builtin_nontemporal_store(c0[r8], orow + n0 + m);
    __builtin_nontemporal_store(c1[r8], orow + n0 + 16 + m);
  }

  // ---- write probs rows in one coalesced non-temporal pass ----
  // 537 MB written once; NT keeps the re-read Q/K/V streams resident in
  // the 192 MB L2 instead of being thrashed by this store stream.
  {
    float* prow = probs + ((size_t)bh * S_ + q0) * S_;
    const int cb0 = t0 * 16, cend = cb0 + ncols;
    for (int i = tid; i < 16 * S_; i += 128) {
      const int r = i >> 11, c = i & (S_ - 1);
      float v = 0.0f;
      if (c >= cb0 && c < cend) v = Sc[r][c - cb0];
      __builtin_nontemporal_store(v, prow + (size_t)r * S_ + c);
    }
  }
}

extern "C" void kernel_launch(void* const* d_in, const int* in_sizes, int n_in,
                              void* d_out, int out_size, void* d_ws, size_t ws_size,
                              hipStream_t stream) {
  const float* Q = (const float*)d_in[0];
  const float* K = (const float*)d_in[1];
  const float* V = (const float*)d_in[2];
  float* ctx   = (float*)d_out;
  float* probs = ctx + (size_t)B_ * H_ * S_ * D_;   // tuple: (context, probs)
  const int nblocks = B_ * H_ * (S_ / 16);          // 4096 workgroups
  sparse_attn_kernel<<<nblocks, 128, 0, stream>>>(Q, K, V, ctx, probs);
}
